// Decoder_34522947125911
// MI455X (gfx1250) — matrix operand order repaired and checked by reading, not compiled, not run on hardware
//
#include <hip/hip_runtime.h>

typedef _Float16 v16h __attribute__((ext_vector_type(16)));
typedef _Float16 v8h  __attribute__((ext_vector_type(8)));
typedef float    v8f  __attribute__((ext_vector_type(8)));

#define B_ 4096
#define T_ 72
#define F_ 9
#define H_ 64

// ---------------------------------------------------------------------------
// WMMA helpers (layouts per CDNA5 ISA)
// ---------------------------------------------------------------------------
__device__ __forceinline__ v8f wmma_f16(v16h a, v16h b, v8f c) {
  // v_wmma_f32_16x16x32_f16
  return __builtin_amdgcn_wmma_f32_16x16x32_f16(false, a, false, b, (short)0, c,
                                                false, false);
}

__device__ __forceinline__ v8f splat8(float v) {
  v8f r;
#pragma unroll
  for (int i = 0; i < 8; ++i) r[i] = v;
  return r;
}

__device__ __forceinline__ float fast_sigmoid(float v) {
  return __builtin_amdgcn_rcpf(1.f + __expf(-v));
}

// A fragment (16x32 f16, MxK): lanes 0-15 hold row M=lane (K 0-7 then 16-23),
// lanes 16-31 hold row M=lane-16 (K 8-15 then 24-31). Source is a row-major
// f16 matrix with leading dimension ldh; kb selects the 32-wide K block.
__device__ __forceinline__ v16h load_a_frag(const _Float16* base, int ldh,
                                            int lane, int kb) {
  int r  = lane & 15;
  int kk = kb + ((lane >> 4) << 3);
  v8h lo = *(const v8h*)(base + r * ldh + kk);
  v8h hi = *(const v8h*)(base + r * ldh + kk + 16);
  v16h a;
#pragma unroll
  for (int i = 0; i < 8; ++i) { a[i] = lo[i]; a[i + 8] = hi[i]; }
  return a;
}

// B fragment: pre-packed; each fragment is 32 lanes x 16 halfs contiguous.
__device__ __forceinline__ v16h load_b_frag(const _Float16* pack, int fragIdx,
                                            int lane) {
  return *(const v16h*)(pack + (size_t)fragIdx * 512 + lane * 16);
}

// ---------------------------------------------------------------------------
// Weight packing: src (K x N, f32 row-major) -> B-fragment f16 layout.
// Fragment index = j*(K/32) + kb. Within a fragment: lane&15 = column,
// lanes 0-15 hold K 0-7 then 16-23, lanes 16-31 hold K 8-15 then 24-31.
// ---------------------------------------------------------------------------
__global__ void pack_b_kernel(const float* __restrict__ src,
                              _Float16* __restrict__ dst, int K, int N) {
  int nkb = K >> 5;
  int nj  = N >> 4;
  int gid = blockIdx.x * blockDim.x + threadIdx.x;
  if (gid >= nkb * nj * 32) return;
  int lane = gid & 31;
  int frag = gid >> 5;
  int kb   = frag % nkb;
  int j    = frag / nkb;
  int col  = j * 16 + (lane & 15);
  int kbase = kb * 32 + ((lane >> 4) << 3);
  _Float16* d = dst + (size_t)frag * 512 + lane * 16;
#pragma unroll
  for (int i = 0; i < 8; ++i) {
    d[i]     = (_Float16)src[(size_t)(kbase + i) * N + col];
    d[i + 8] = (_Float16)src[(size_t)(kbase + 16 + i) * N + col];
  }
}

// ---------------------------------------------------------------------------
// Fused bidirectional GRU scan (reset_after=True).
//  - block = 128 threads = 4 waves; wave w owns output columns [16w, 16w+16)
//  - blockIdx.x = 16-row batch tile, blockIdx.y = direction (0 fwd, 1 bwd)
//  - hidden state: f32 C-fragment in registers; f16 copy in LDS provides the
//    A operand each step (16x64 -> two 16x32 A fragments)
//  - z and r pre-activations share one WMMA chain seeded with bi+br
//  - packed weights staged in LDS; a compiler barrier each step prevents
//    LICM from pinning weight fragments in VGPRs (keeps usage < 256)
//  - L1: x@W (K=9) fused in VALU from LDS-staged x tile; stores seq f16
//  - L2: seq@W2 (K=128) fused via 12 extra WMMAs/step/wave from global f16
// ---------------------------------------------------------------------------
template <int ACT, bool L1>
__global__ __launch_bounds__(128) void gru_scan_kernel(
    const float* __restrict__ x, const _Float16* __restrict__ seqin,
    const _Float16* __restrict__ Upf, const _Float16* __restrict__ Upb,
    const _Float16* __restrict__ Wpf, const _Float16* __restrict__ Wpb,
    const float* __restrict__ W1f, const float* __restrict__ W1b,
    const float* __restrict__ bif, const float* __restrict__ bib,
    const float* __restrict__ brf, const float* __restrict__ brb,
    const float* __restrict__ h0f, const float* __restrict__ h0b,
    _Float16* __restrict__ seqout, float* __restrict__ feat) {
  __shared__ _Float16 hbuf[16 * 64];
  __shared__ float xbuf[16 * F_];
  __shared__ _Float16 ulds[24 * 512];                  // packed U (24 KB)
  __shared__ _Float16 wlds[L1 ? 16 : 48 * 512];        // packed W2 (48 KB, L2)

  const int tid  = threadIdx.x;
  const int wv   = tid >> 5;   // output column group (j = wv)
  const int lane = tid & 31;
  const int hi   = lane >> 4;
  const int l15  = lane & 15;
  const int tile = blockIdx.x;
  const int dir  = blockIdx.y;
  const int b0   = tile * 16;

  const _Float16* Up = dir ? Upb : Upf;
  const _Float16* Wp = dir ? Wpb : Wpf;
  const float* W1 = dir ? W1b : W1f;
  const float* bi = dir ? bib : bif;
  const float* br = dir ? brb : brf;
  const float* h0 = dir ? h0b : h0f;

  // Stage packed weights into LDS (vectorized 16B copies).
  {
    const uint4* s = (const uint4*)Up;
    uint4* d = (uint4*)ulds;
    for (int i = tid; i < 24 * 512 / 8; i += 128) d[i] = s[i];
  }
  if constexpr (!L1) {
    const uint4* s = (const uint4*)Wp;
    uint4* d = (uint4*)wlds;
    for (int i = tid; i < 48 * 512 / 8; i += 128) d[i] = s[i];
  }

  // Stage initial hidden state (f16 in LDS for the A operand, f32 in regs).
  for (int idx = tid; idx < 16 * 64; idx += 128) {
    float v = 0.f;
    if constexpr (L1) {
      int rr = idx >> 6, cc = idx & 63;
      v = h0[(size_t)(b0 + rr) * H_ + cc];
    }
    hbuf[idx] = (_Float16)v;
  }
  v8f h;
#pragma unroll
  for (int e = 0; e < 8; ++e) {
    float v = 0.f;
    if constexpr (L1) v = h0[(size_t)(b0 + e + 8 * hi) * H_ + 16 * wv + l15];
    h[e] = v;
  }

  // Loop-invariant per-lane biases; combine bi+br for the z/r chains.
  const int cz = 16 * wv + l15;
  const int cr = 16 * (wv + 4) + l15;
  const int ch = 16 * (wv + 8) + l15;
  const float bZ  = bi[cz] + br[cz];
  const float bR  = bi[cr] + br[cr];
  const float bXh = bi[ch];
  const float bRh = br[ch];

  float wcz[F_], wcr[F_], wch[F_];
  if constexpr (L1) {
#pragma unroll
    for (int f = 0; f < F_; ++f) {
      wcz[f] = W1[f * 192 + cz];
      wcr[f] = W1[f * 192 + cr];
      wch[f] = W1[f * 192 + ch];
    }
  }

  if constexpr (L1) {  // stage x tile for step 0
    int t0 = dir ? (T_ - 1) : 0;
    for (int idx = tid; idx < 16 * F_; idx += 128) {
      int rr = idx / F_, ff = idx % F_;
      xbuf[idx] = x[((size_t)(b0 + rr) * T_ + t0) * F_ + ff];
    }
  }
  __syncthreads();

  for (int s = 0; s < T_; ++s) {
    // Compiler barrier: keep weight-fragment loads inside the loop (LDS reads
    // per step) instead of hoisted into hundreds of VGPRs.
    asm volatile("" ::: "memory");
    int t = dir ? (T_ - 1 - s) : s;

    // A fragments of h (K=64) from LDS.
    v16h a0 = load_a_frag(hbuf, 64, lane, 0);
    v16h a1 = load_a_frag(hbuf, 64, lane, 32);

    v8f accZ  = splat8(bZ);   // xz + rz (+ biases)
    v8f accR  = splat8(bR);   // xr + rr (+ biases)
    v8f accXh = splat8(bXh);  // xh + bi
    v8f accRh = splat8(bRh);  // rh + br

    if constexpr (!L1) {
      // Fused input projection: seq_t (16x128 f16, global) @ W2 (LDS).
      const _Float16* abase = seqin + ((size_t)b0 * T_ + t) * 128;
      v16h a2[4];
#pragma unroll
      for (int kb = 0; kb < 4; ++kb)
        a2[kb] = load_a_frag(abase, T_ * 128, lane, kb * 32);
#pragma unroll
      for (int kb = 0; kb < 4; ++kb)
        accZ = wmma_f16(a2[kb], load_b_frag(wlds, wv * 4 + kb, lane), accZ);
#pragma unroll
      for (int kb = 0; kb < 4; ++kb)
        accR = wmma_f16(a2[kb], load_b_frag(wlds, (wv + 4) * 4 + kb, lane), accR);
#pragma unroll
      for (int kb = 0; kb < 4; ++kb)
        accXh = wmma_f16(a2[kb], load_b_frag(wlds, (wv + 8) * 4 + kb, lane), accXh);
    } else {
      // Fused input projection: x_t (16x9 f32, LDS) @ W1 columns (VALU).
#pragma unroll
      for (int e = 0; e < 8; ++e) {
        float xr_[F_];
#pragma unroll
        for (int f = 0; f < F_; ++f) xr_[f] = xbuf[(e + 8 * hi) * F_ + f];
        float az = 0.f, ar = 0.f, ah = 0.f;
#pragma unroll
        for (int f = 0; f < F_; ++f) {
          az += xr_[f] * wcz[f];
          ar += xr_[f] * wcr[f];
          ah += xr_[f] * wch[f];
        }
        accZ[e] += az;
        accR[e] += ar;
        accXh[e] += ah;
      }
    }

    // Recurrent projections from LDS-resident packed U.
    accZ  = wmma_f16(a0, load_b_frag(ulds, wv * 2 + 0, lane), accZ);
    accZ  = wmma_f16(a1, load_b_frag(ulds, wv * 2 + 1, lane), accZ);
    accR  = wmma_f16(a0, load_b_frag(ulds, (wv + 4) * 2 + 0, lane), accR);
    accR  = wmma_f16(a1, load_b_frag(ulds, (wv + 4) * 2 + 1, lane), accR);
    accRh = wmma_f16(a0, load_b_frag(ulds, (wv + 8) * 2 + 0, lane), accRh);
    accRh = wmma_f16(a1, load_b_frag(ulds, (wv + 8) * 2 + 1, lane), accRh);

    // Gates.
#pragma unroll
    for (int e = 0; e < 8; ++e) {
      float z   = fast_sigmoid(accZ[e]);
      float r   = fast_sigmoid(accR[e]);
      float pre = accXh[e] + r * accRh[e];
      float hh;
      if (ACT == 0) {
        hh = fmaxf(pre, 0.f);
      } else {
        float pc = fminf(fmaxf(pre, -15.f), 15.f);
        float t2 = __expf(2.f * pc);
        hh = (t2 - 1.f) * __builtin_amdgcn_rcpf(t2 + 1.f);
      }
      h[e] = z * h[e] + (1.f - z) * hh;
    }

    __syncthreads();  // everyone done reading hbuf/xbuf

    // Write new hidden state (this wave's 16 columns) to LDS as f16.
#pragma unroll
    for (int e = 0; e < 8; ++e)
      hbuf[(e + 8 * hi) * 64 + 16 * wv + l15] = (_Float16)h[e];

    if constexpr (L1) {  // stage x tile for next step
      if (s + 1 < T_) {
        int tn = dir ? (T_ - 2 - s) : (s + 1);
        for (int idx = tid; idx < 16 * F_; idx += 128) {
          int rr = idx / F_, ff = idx % F_;
          xbuf[idx] = x[((size_t)(b0 + rr) * T_ + tn) * F_ + ff];
        }
      }
    }
    __syncthreads();  // hbuf complete

    if constexpr (L1) {  // coalesced f16 sequence store (b128 per thread)
      if (tid < 64) {
        int row = tid >> 2, chk = tid & 3;
        v8h v = *(const v8h*)(hbuf + row * 64 + chk * 8);
        *(v8h*)(seqout + ((size_t)(b0 + row) * T_ + t) * 128 + dir * 64 +
                chk * 8) = v;
      }
    }
  }

  if constexpr (!L1) {  // final state -> feat (B,128) f32
#pragma unroll
    for (int e = 0; e < 8; ++e)
      feat[(size_t)(b0 + e + 8 * hi) * 128 + dir * 64 + 16 * wv + l15] = h[e];
  }
}

// ---------------------------------------------------------------------------
// Dense(128, relu) -> Dense(24) -> softmax. One block per batch row.
// ---------------------------------------------------------------------------
__global__ __launch_bounds__(128) void head_kernel(
    const float* __restrict__ feat, const float* __restrict__ dW,
    const float* __restrict__ db, const float* __restrict__ oW,
    const float* __restrict__ ob, float* __restrict__ out) {
  __shared__ float sf[128];
  __shared__ float sd[128];
  __shared__ float sl[24];
  __shared__ float sred[2];
  int row = blockIdx.x, tid = threadIdx.x;
  sf[tid] = feat[(size_t)row * 128 + tid];
  __syncthreads();
  float acc = db[tid];
  for (int k = 0; k < 128; ++k) acc += sf[k] * dW[k * 128 + tid];
  sd[tid] = fmaxf(acc, 0.f);
  __syncthreads();
  if (tid < 24) {
    float s = ob[tid];
    for (int k = 0; k < 128; ++k) s += sd[k] * oW[k * 24 + tid];
    sl[tid] = s;
  }
  __syncthreads();
  if (tid == 0) {
    float m = sl[0];
    for (int i = 1; i < 24; ++i) m = fmaxf(m, sl[i]);
    float ssum = 0.f;
    for (int i = 0; i < 24; ++i) ssum += __expf(sl[i] - m);
    sred[0] = m;
    sred[1] = ssum;
  }
  __syncthreads();
  if (tid < 24) out[(size_t)row * 24 + tid] = __expf(sl[tid] - sred[0]) / sred[1];
}

// ---------------------------------------------------------------------------
extern "C" void kernel_launch(void* const* d_in, const int* in_sizes, int n_in,
                              void* d_out, int out_size, void* d_ws,
                              size_t ws_size, hipStream_t stream) {
  (void)in_sizes; (void)n_in; (void)out_size;
  const float* x      = (const float*)d_in[0];
  const float* h0f    = (const float*)d_in[1];
  const float* h0b    = (const float*)d_in[2];
  const float* d1f_W  = (const float*)d_in[3];
  const float* d1f_U  = (const float*)d_in[4];
  const float* d1f_bi = (const float*)d_in[5];
  const float* d1f_br = (const float*)d_in[6];
  const float* d1b_W  = (const float*)d_in[7];
  const float* d1b_U  = (const float*)d_in[8];
  const float* d1b_bi = (const float*)d_in[9];
  const float* d1b_br = (const float*)d_in[10];
  const float* d2f_W  = (const float*)d_in[11];
  const float* d2f_U  = (const float*)d_in[12];
  const float* d2f_bi = (const float*)d_in[13];
  const float* d2f_br = (const float*)d_in[14];
  const float* d2b_W  = (const float*)d_in[15];
  const float* d2b_U  = (const float*)d_in[16];
  const float* d2b_bi = (const float*)d_in[17];
  const float* d2b_br = (const float*)d_in[18];
  const float* dW     = (const float*)d_in[19];
  const float* db     = (const float*)d_in[20];
  const float* oW     = (const float*)d_in[21];
  const float* ob     = (const float*)d_in[22];
  float* out = (float*)d_out;

  char* ws = (char*)d_ws;
  size_t off = 0;
  auto alloc = [&](size_t bytes) {
    size_t o = off;
    off = (off + bytes + 255) & ~(size_t)255;
    return o;
  };
  size_t oU1f = alloc(24 * 512 * 2), oU1b = alloc(24 * 512 * 2);
  size_t oU2f = alloc(24 * 512 * 2), oU2b = alloc(24 * 512 * 2);
  size_t oW2f = alloc(48 * 512 * 2), oW2b = alloc(48 * 512 * 2);
  size_t oSeq  = alloc((size_t)B_ * T_ * 128 * 2);
  size_t oFeat = alloc((size_t)B_ * 128 * 4);
  if (off > ws_size) return;

  _Float16* U1fp = (_Float16*)(ws + oU1f);
  _Float16* U1bp = (_Float16*)(ws + oU1b);
  _Float16* U2fp = (_Float16*)(ws + oU2f);
  _Float16* U2bp = (_Float16*)(ws + oU2b);
  _Float16* W2fp = (_Float16*)(ws + oW2f);
  _Float16* W2bp = (_Float16*)(ws + oW2b);
  _Float16* seqp = (_Float16*)(ws + oSeq);
  float* featp   = (float*)(ws + oFeat);

  // Pack recurrent / layer-2 input weights into B-fragment f16 layout.
  pack_b_kernel<<<3, 256, 0, stream>>>(d1f_U, U1fp, 64, 192);
  pack_b_kernel<<<3, 256, 0, stream>>>(d1b_U, U1bp, 64, 192);
  pack_b_kernel<<<3, 256, 0, stream>>>(d2f_U, U2fp, 64, 192);
  pack_b_kernel<<<3, 256, 0, stream>>>(d2b_U, U2bp, 64, 192);
  pack_b_kernel<<<6, 256, 0, stream>>>(d2f_W, W2fp, 128, 192);
  pack_b_kernel<<<6, 256, 0, stream>>>(d2b_W, W2bp, 128, 192);

  dim3 sgrid(B_ / 16, 2);
  // Layer 1: relu GRU, stores sequence; both directions in one launch.
  gru_scan_kernel<0, true><<<sgrid, 128, 0, stream>>>(
      x, nullptr, U1fp, U1bp, nullptr, nullptr, d1f_W, d1b_W, d1f_bi, d1b_bi,
      d1f_br, d1b_br, h0f, h0b, seqp, nullptr);
  // Layer 2: tanh GRU, fused seq@W2, stores final states.
  gru_scan_kernel<1, false><<<sgrid, 128, 0, stream>>>(
      nullptr, seqp, U2fp, U2bp, W2fp, W2bp, nullptr, nullptr, d2f_bi, d2b_bi,
      d2f_br, d2b_br, nullptr, nullptr, nullptr, featp);

  head_kernel<<<B_, 128, 0, stream>>>(featp, dW, db, oW, ob, out);
}
